// SASparseResNet3D_72962904425004
// MI455X (gfx1250) — matrix operand order repaired and checked by reading, not compiled, not run on hardware
//
#include <hip/hip_runtime.h>

// ---------------------------------------------------------------------------
// SASparseResNet3D on gfx1250 (CDNA5): WMMA f16 implicit-GEMM pipeline
// B=2, CIN=32, CH=64, PRED=32, D=64 -> spatial out 32^3, NTOT = 2*32768
// v2: branchless gathers, packed ds_store_b64 LDS fills, K=64 phases with
//     2 N-accumulators per wave -> 4 v_wmma per wave per barrier phase.
// ---------------------------------------------------------------------------

#define NS   32768      // 32*32*32
#define NTOT 65536      // B * NS

typedef __attribute__((ext_vector_type(16))) _Float16 v16h;
typedef __attribute__((ext_vector_type(8)))  _Float16 v8h;
typedef __attribute__((ext_vector_type(4)))  _Float16 v4h;
typedef __attribute__((ext_vector_type(8)))  float    v8f;

__device__ __forceinline__ float lrelu(float v) { return v > 0.f ? v : 0.01f * v; }

__device__ __forceinline__ v8f wmma16(v16h A, v16h B, v8f C) {
  return __builtin_amdgcn_wmma_f32_16x16x32_f16(false, A, false, B, (short)0, C,
                                                false, false);
}

// A fragment, contiguous k order: lane = (m%16) + 16*(k>>4);
// per lane the 16 halves are one contiguous run of 16 k-values -> 2x b128 loads.
__device__ __forceinline__ v16h load_afrag(const _Float16* __restrict__ W,
                                           int mrow, int Kdim, int kc, int g) {
  const _Float16* p = W + (size_t)mrow * Kdim + kc + 16 * g;
  v8h a0 = *(const v8h*)p;
  v8h a1 = *(const v8h*)(p + 8);
  v16h A;
#pragma unroll
  for (int i = 0; i < 8; ++i) { A[i] = a0[i]; A[i + 8] = a1[i]; }
  return A;
}

// ---------------------------------------------------------------------------
// Weight conversion f32 -> f16, reordered to [M][K] with K-chunk = one tap
// ---------------------------------------------------------------------------
__global__ __launch_bounds__(256) void k_wcvt(
    const float* __restrict__ dw, const float* __restrict__ w1s,
    const float* __restrict__ w2s, const float* __restrict__ w3s,
    const float* __restrict__ wf,
    _Float16* __restrict__ Wd, _Float16* __restrict__ W1,
    _Float16* __restrict__ W2, _Float16* __restrict__ W3,
    _Float16* __restrict__ Wf) {
  const int id = blockIdx.y;
  int Ne;
  switch (id) {
    case 0: Ne = 64 * 2048; break;       // down: [64][tap*32+ci]
    case 1: Ne = 3 * 64 * 64; break;     // w1s passthrough
    case 2: Ne = 3 * 64 * 1728; break;   // w2s: [i][co][tap*64+ci]
    case 3: Ne = 3 * 64 * 64; break;     // w3s passthrough
    default: Ne = 32 * 64; break;        // wf passthrough
  }
  for (int e = blockIdx.x * 256 + threadIdx.x; e < Ne; e += gridDim.x * 256) {
    if (id == 0) {
      int co = e >> 11, k = e & 2047, tap = k >> 5, ci = k & 31;
      Wd[e] = (_Float16)dw[(co * 32 + ci) * 64 + tap];
    } else if (id == 1) {
      W1[e] = (_Float16)w1s[e];
    } else if (id == 2) {
      int i = e / 110592, r = e % 110592, co = r / 1728, k = r % 1728;
      int tap = k >> 6, ci = k & 63;
      W2[e] = (_Float16)w2s[(((i * 64 + co) * 64 + ci) * 27) + tap];
    } else if (id == 3) {
      W3[e] = (_Float16)w3s[e];
    } else {
      Wf[e] = (_Float16)wf[e];
    }
  }
}

// ---------------------------------------------------------------------------
// Mask prep: mc0 = 64/clip(4^3 count, stride2 pad1), m = maxpool2 > 0.5
// m also written to d_out tail (second tuple output).
// ---------------------------------------------------------------------------
__global__ __launch_bounds__(256) void k_maskprep(
    const float* __restrict__ mask, float* __restrict__ mc0,
    float* __restrict__ mout, float* __restrict__ m_dout) {
  const int n = blockIdx.x * 256 + threadIdx.x;
  const int b = n >> 15, v3 = n & 32767;
  const int od = v3 >> 10, oh = (v3 >> 5) & 31, ow = v3 & 31;
  const float* mb = mask + (size_t)b * 262144;
  float cnt = 0.f;
#pragma unroll
  for (int kd = 0; kd < 4; ++kd) {
    const int id = 2 * od - 1 + kd;
    const float sz = ((unsigned)id < 64u) ? 1.f : 0.f;
#pragma unroll
    for (int kh = 0; kh < 4; ++kh) {
      const int ih = 2 * oh - 1 + kh;
      const float sy = ((unsigned)ih < 64u) ? sz : 0.f;
#pragma unroll
      for (int kw = 0; kw < 4; ++kw) {
        const int iw = 2 * ow - 1 + kw;
        const float sx = ((unsigned)iw < 64u) ? sy : 0.f;
        cnt += mb[(id & 63) * 4096 + (ih & 63) * 64 + (iw & 63)] * sx;
      }
    }
  }
  mc0[n] = 64.f / fmaxf(cnt, 1e-5f);
  float mx = -1e30f;
#pragma unroll
  for (int dz = 0; dz < 2; ++dz)
#pragma unroll
    for (int dy = 0; dy < 2; ++dy)
#pragma unroll
      for (int dx = 0; dx < 2; ++dx)
        mx = fmaxf(mx, mb[(2 * od + dz) * 4096 + (2 * oh + dy) * 64 + (2 * ow + dx)]);
  const float mv = mx > 0.5f ? 1.f : 0.f;
  mout[n] = mv;
  m_dout[n] = mv;
}

// mc = 27 / clip(3x3x3 ones-conv of m, pad 1)
__global__ __launch_bounds__(256) void k_mc(const float* __restrict__ m,
                                            float* __restrict__ mc) {
  const int n = blockIdx.x * 256 + threadIdx.x;
  const int b = n >> 15, v3 = n & 32767;
  const int od = v3 >> 10, oh = (v3 >> 5) & 31, ow = v3 & 31;
  const float* mb = m + (size_t)b * NS;
  float cnt = 0.f;
#pragma unroll
  for (int dz = -1; dz <= 1; ++dz) {
    const int zz = od + dz;
    const float sz = ((unsigned)zz < 32u) ? 1.f : 0.f;
#pragma unroll
    for (int dy = -1; dy <= 1; ++dy) {
      const int yy = oh + dy;
      const float sy = ((unsigned)yy < 32u) ? sz : 0.f;
#pragma unroll
      for (int dx = -1; dx <= 1; ++dx) {
        const int xx = ow + dx;
        const float sx = ((unsigned)xx < 32u) ? sy : 0.f;
        cnt += mb[((zz & 31) << 10) + ((yy & 31) << 5) + (xx & 31)] * sx;
      }
    }
  }
  mc[n] = 27.f / fmaxf(cnt, 1e-5f);
}

// ---------------------------------------------------------------------------
// Downsample conv as implicit GEMM: M=64, N=NTOT, K=2048 (64 taps x 32 cin)
// 32 phases of K=64 (2 taps); N tile 64; 4 v_wmma per wave per phase.
// epilogue: x = (acc + db[M]) * mc0[n]
// ---------------------------------------------------------------------------
__global__ __launch_bounds__(256) void k_downconv(
    const _Float16* __restrict__ W, const float* __restrict__ feat,
    const float* __restrict__ db, const float* __restrict__ mc0,
    float* __restrict__ xout) {
  __shared__ __align__(32) _Float16 Bt[64 * 72];
  const int tid = threadIdx.x;
  const int wave = tid >> 5, lane = tid & 31;
  const int wm = wave & 3, wc = wave >> 2;   // 4 M-tiles x 2 N-columns
  const int g = lane >> 4, col = lane & 15;
  const int nbase = blockIdx.x * 64;

  // loop-invariant gather decode: one voxel per thread
  const int nl = tid & 63;
  const int n = nbase + nl;
  const int b = n >> 15, v3 = n & 32767;
  const int od2 = 2 * (v3 >> 10) - 1;
  const int oh2 = 2 * ((v3 >> 5) & 31) - 1;
  const int ow2 = 2 * (v3 & 31) - 1;
  const float* fb = feat + (size_t)b * 32 * 262144;

  v8f acc0 = {}, acc1 = {};
  for (int ph = 0; ph < 32; ++ph) {
    // two taps per phase; branchless clamp + 0/1 selector
    int soff[2]; float sel[2];
#pragma unroll
    for (int t = 0; t < 2; ++t) {
      const int tap = ph * 2 + t;
      const int id = od2 + (tap >> 4);
      const int ih = oh2 + ((tap >> 2) & 3);
      const int iw = ow2 + (tap & 3);
      sel[t] = ((unsigned)id < 64u && (unsigned)ih < 64u && (unsigned)iw < 64u)
                   ? 1.f : 0.f;
      soff[t] = (id & 63) * 4096 + (ih & 63) * 64 + (iw & 63);
    }
    __syncthreads();
#pragma unroll
    for (int j = 0; j < 4; ++j) {
      const int kq = (tid >> 6) + 4 * j;     // 0..15
      const int t = kq >> 3;                 // tap slot in phase
      const int ci0 = (kq & 7) * 4;          // 0..28
      const float* p = fb + (size_t)ci0 * 262144 + soff[t];
      v4h pk;
#pragma unroll
      for (int jj = 0; jj < 4; ++jj)
        pk[jj] = (_Float16)(p[(size_t)jj * 262144] * sel[t]);
      *(v4h*)(Bt + nl * 72 + t * 32 + ci0) = pk;
    }
    __syncthreads();
    const int kc = ph * 64;
    const v16h A0 = load_afrag(W, wm * 16 + col, 2048, kc, g);
    const v16h A1 = load_afrag(W, wm * 16 + col, 2048, kc + 32, g);
    const _Float16* r0 = Bt + (wc * 32 + col) * 72 + 16 * g;
    const _Float16* r1 = Bt + (wc * 32 + 16 + col) * 72 + 16 * g;
    acc0 = wmma16(A0, *(const v16h*)r0, acc0);
    acc0 = wmma16(A1, *(const v16h*)(r0 + 32), acc0);
    acc1 = wmma16(A0, *(const v16h*)r1, acc1);
    acc1 = wmma16(A1, *(const v16h*)(r1 + 32), acc1);
  }
#pragma unroll
  for (int r = 0; r < 8; ++r) {
    const int M = wm * 16 + r + 8 * g;
    const int n0 = nbase + wc * 32 + col;
    const int n1 = n0 + 16;
    xout[(size_t)M * NTOT + n0] = (acc0[r] + db[M]) * mc0[n0];
    xout[(size_t)M * NTOT + n1] = (acc1[r] + db[M]) * mc0[n1];
  }
}

// ---------------------------------------------------------------------------
// BatchNorm: per-channel stats (biased var) -> scale/shift, then apply+lrelu
// ---------------------------------------------------------------------------
__global__ __launch_bounds__(256) void k_bnstats(
    const float* __restrict__ x, const float* __restrict__ gamma,
    const float* __restrict__ beta, float* __restrict__ scale,
    float* __restrict__ shift) {
  const int c = blockIdx.x;
  __shared__ float ss[256], sq[256];
  const float4* xv = (const float4*)(x + (size_t)c * NTOT);
  float s = 0.f, q = 0.f;
  for (int i = threadIdx.x; i < NTOT / 4; i += 256) {
    float4 v = xv[i];
    s += v.x + v.y + v.z + v.w;
    q += v.x * v.x + v.y * v.y + v.z * v.z + v.w * v.w;
  }
  ss[threadIdx.x] = s; sq[threadIdx.x] = q;
  __syncthreads();
  for (int o = 128; o > 0; o >>= 1) {
    if (threadIdx.x < o) {
      ss[threadIdx.x] += ss[threadIdx.x + o];
      sq[threadIdx.x] += sq[threadIdx.x + o];
    }
    __syncthreads();
  }
  if (threadIdx.x == 0) {
    float mean = ss[0] * (1.f / NTOT);
    float var = sq[0] * (1.f / NTOT) - mean * mean;
    float sc = gamma[c] * rsqrtf(var + 1e-5f);
    scale[c] = sc;
    shift[c] = beta[c] - mean * sc;
  }
}

__global__ __launch_bounds__(256) void k_bnapply(float4* __restrict__ x,
                                                 const float* __restrict__ scale,
                                                 const float* __restrict__ shift) {
  const int i = blockIdx.x * 256 + threadIdx.x;  // 64*NTOT/4 threads exactly
  const int c = i >> 14;                         // (4i) >> 16
  const float sc = scale[c], sh = shift[c];
  float4 v = x[i];
  v.x = lrelu(v.x * sc + sh);
  v.y = lrelu(v.y * sc + sh);
  v.z = lrelu(v.z * sc + sh);
  v.w = lrelu(v.w * sc + sh);
  x[i] = v;
}

// ---------------------------------------------------------------------------
// 1x1x1 conv GEMM: K=64 in one phase, 2 N-accs per wave (4 v_wmma).
//   MODE 1: h1 = lrelu((W@(x*m)+b)*m)
//   MODE 3: x  = lrelu(xold + (W@h2+b)*m*mc)  (in place)
//   MODE 4: final CH->PRED, write NCDHW into d_out
// ---------------------------------------------------------------------------
template <int MTILES, int MODE>
__global__ __launch_bounds__(256) void k_gemm1x1(
    const _Float16* __restrict__ W, const float* __restrict__ Xin,
    const float* __restrict__ bias, const float* __restrict__ mk,
    const float* __restrict__ mc, const float* xold, float* out) {
  constexpr int NT = (8 / MTILES) * 32;  // N tile: 64 (M=64) / 128 (M=32)
  __shared__ __align__(32) _Float16 Bt[NT * 72];
  const int tid = threadIdx.x;
  const int wave = tid >> 5, lane = tid & 31;
  const int wm = wave % MTILES, wc = wave / MTILES;
  const int g = lane >> 4, col = lane & 15;
  const int nbase = blockIdx.x * NT;

  const int nl = tid & (NT - 1);
  const int n = nbase + nl;
  float msel = 1.f;
  if constexpr (MODE == 1) msel = mk[n];

#pragma unroll
  for (int j = 0; j < NT / 16; ++j) {
    const int kq = (tid / NT) + (256 / NT) * j;  // 0..15
    const int ci0 = kq * 4;
    v4h pk;
#pragma unroll
    for (int jj = 0; jj < 4; ++jj)
      pk[jj] = (_Float16)(Xin[(size_t)(ci0 + jj) * NTOT + n] * msel);
    *(v4h*)(Bt + nl * 72 + ci0) = pk;
  }
  __syncthreads();
  const v16h A0 = load_afrag(W, wm * 16 + col, 64, 0, g);
  const v16h A1 = load_afrag(W, wm * 16 + col, 64, 32, g);
  const _Float16* r0 = Bt + (wc * 32 + col) * 72 + 16 * g;
  const _Float16* r1 = Bt + (wc * 32 + 16 + col) * 72 + 16 * g;
  v8f acc0 = {}, acc1 = {};
  acc0 = wmma16(A0, *(const v16h*)r0, acc0);
  acc0 = wmma16(A1, *(const v16h*)(r0 + 32), acc0);
  acc1 = wmma16(A0, *(const v16h*)r1, acc1);
  acc1 = wmma16(A1, *(const v16h*)(r1 + 32), acc1);

#pragma unroll
  for (int a = 0; a < 2; ++a) {
    const v8f acc = a ? acc1 : acc0;
#pragma unroll
    for (int r = 0; r < 8; ++r) {
      const int M = wm * 16 + r + 8 * g;
      const int nn = nbase + wc * 32 + a * 16 + col;
      float v = acc[r] + bias[M];
      if constexpr (MODE == 1) {
        out[(size_t)M * NTOT + nn] = lrelu(v * mk[nn]);
      } else if constexpr (MODE == 3) {
        v = v * mk[nn] * mc[nn];
        out[(size_t)M * NTOT + nn] = lrelu(xold[(size_t)M * NTOT + nn] + v);
      } else {  // MODE 4: NCDHW output [b][p][voxel]
        const int bb = nn >> 15, vox = nn & 32767;
        out[(size_t)bb * (32 * NS) + (size_t)M * NS + vox] = v;
      }
    }
  }
}

// ---------------------------------------------------------------------------
// 3x3x3 conv GEMM: M=64, K=1728; 27 phases of K=64 (one tap, 64 cin).
// epilogue: h2 = lrelu((acc + b2[M]) * m[n])
// ---------------------------------------------------------------------------
__global__ __launch_bounds__(256) void k_conv3x3(
    const _Float16* __restrict__ W, const float* __restrict__ Xin,
    const float* __restrict__ bias, const float* __restrict__ mk,
    float* __restrict__ out) {
  __shared__ __align__(32) _Float16 Bt[64 * 72];
  const int tid = threadIdx.x;
  const int wave = tid >> 5, lane = tid & 31;
  const int wm = wave & 3, wc = wave >> 2;
  const int g = lane >> 4, col = lane & 15;
  const int nbase = blockIdx.x * 64;

  const int nl = tid & 63;
  const int n = nbase + nl;
  const int b = n >> 15, v3 = n & 32767;
  const int od = v3 >> 10, oh = (v3 >> 5) & 31, ow = v3 & 31;
  const float* xb = Xin + (size_t)(b << 15);

  v8f acc0 = {}, acc1 = {};
  for (int tap = 0; tap < 27; ++tap) {
    const int zz = od + tap / 9 - 1;
    const int yy = oh + (tap / 3) % 3 - 1;
    const int xx = ow + tap % 3 - 1;
    const float sel =
        ((unsigned)zz < 32u && (unsigned)yy < 32u && (unsigned)xx < 32u) ? 1.f : 0.f;
    const int soff = ((zz & 31) << 10) + ((yy & 31) << 5) + (xx & 31);
    __syncthreads();
#pragma unroll
    for (int j = 0; j < 4; ++j) {
      const int ci0 = ((tid >> 6) + 4 * j) * 4;  // 0..60
      const float* p = xb + (size_t)ci0 * NTOT + soff;
      v4h pk;
#pragma unroll
      for (int jj = 0; jj < 4; ++jj)
        pk[jj] = (_Float16)(p[(size_t)jj * NTOT] * sel);
      *(v4h*)(Bt + nl * 72 + ci0) = pk;
    }
    __syncthreads();
    const int kc = tap * 64;
    const v16h A0 = load_afrag(W, wm * 16 + col, 1728, kc, g);
    const v16h A1 = load_afrag(W, wm * 16 + col, 1728, kc + 32, g);
    const _Float16* r0 = Bt + (wc * 32 + col) * 72 + 16 * g;
    const _Float16* r1 = Bt + (wc * 32 + 16 + col) * 72 + 16 * g;
    acc0 = wmma16(A0, *(const v16h*)r0, acc0);
    acc0 = wmma16(A1, *(const v16h*)(r0 + 32), acc0);
    acc1 = wmma16(A0, *(const v16h*)r1, acc1);
    acc1 = wmma16(A1, *(const v16h*)(r1 + 32), acc1);
  }
#pragma unroll
  for (int r = 0; r < 8; ++r) {
    const int M = wm * 16 + r + 8 * g;
    const int n0 = nbase + wc * 32 + col;
    const int n1 = n0 + 16;
    out[(size_t)M * NTOT + n0] = lrelu((acc0[r] + bias[M]) * mk[n0]);
    out[(size_t)M * NTOT + n1] = lrelu((acc1[r] + bias[M]) * mk[n1]);
  }
}

// ---------------------------------------------------------------------------
extern "C" void kernel_launch(void* const* d_in, const int* in_sizes, int n_in,
                              void* d_out, int out_size, void* d_ws, size_t ws_size,
                              hipStream_t stream) {
  const float* feat  = (const float*)d_in[0];
  const float* mask  = (const float*)d_in[1];
  const float* dw    = (const float*)d_in[2];
  const float* db    = (const float*)d_in[3];
  const float* gamma = (const float*)d_in[4];
  const float* beta  = (const float*)d_in[5];
  const float* w1s   = (const float*)d_in[6];
  const float* b1s   = (const float*)d_in[7];
  const float* w2s   = (const float*)d_in[8];
  const float* b2s   = (const float*)d_in[9];
  const float* w3s   = (const float*)d_in[10];
  const float* b3s   = (const float*)d_in[11];
  const float* wf    = (const float*)d_in[12];
  const float* bf    = (const float*)d_in[13];
  float* out = (float*)d_out;

  char* ws = (char*)d_ws;
  float*    mc0   = (float*)(ws + 0x000000);   // 256 KB
  float*    mbuf  = (float*)(ws + 0x040000);   // 256 KB
  float*    mcbuf = (float*)(ws + 0x080000);   // 256 KB
  float*    scale = (float*)(ws + 0x0C0000);
  float*    shift = (float*)(ws + 0x0C0400);
  _Float16* Wd    = (_Float16*)(ws + 0x100000);  // 64*2048 halves
  _Float16* W1    = (_Float16*)(ws + 0x140000);  // 3*64*64
  _Float16* W2    = (_Float16*)(ws + 0x150000);  // 3*64*1728
  _Float16* W3    = (_Float16*)(ws + 0x200000);  // 3*64*64
  _Float16* Wf    = (_Float16*)(ws + 0x210000);  // 32*64
  float*    x     = (float*)(ws + 0x0400000);    // 16 MB  [64][NTOT]
  float*    h1    = (float*)(ws + 0x1400000);    // 16 MB
  float*    h2    = (float*)(ws + 0x2400000);    // 16 MB

  // weights -> f16 WMMA layout
  k_wcvt<<<dim3(512, 5), 256, 0, stream>>>(dw, w1s, w2s, w3s, wf, Wd, W1, W2, W3, Wf);
  // masks; m also goes to d_out tail (tuple second output)
  k_maskprep<<<NTOT / 256, 256, 0, stream>>>(mask, mc0, mbuf, out + 2 * 32 * NS);
  k_mc<<<NTOT / 256, 256, 0, stream>>>(mbuf, mcbuf);
  // downsample conv (WMMA) with mask-count renorm
  k_downconv<<<NTOT / 64, 256, 0, stream>>>(Wd, feat, db, mc0, x);
  // batchnorm + lrelu
  k_bnstats<<<64, 256, 0, stream>>>(x, gamma, beta, scale, shift);
  k_bnapply<<<(64 * NTOT / 4) / 256, 256, 0, stream>>>((float4*)x, scale, shift);
  // three masked residual blocks
  for (int i = 0; i < 3; ++i) {
    k_gemm1x1<4, 1><<<NTOT / 64, 256, 0, stream>>>(W1 + i * 4096, x, b1s + i * 64,
                                                   mbuf, nullptr, nullptr, h1);
    k_conv3x3<<<NTOT / 64, 256, 0, stream>>>(W2 + i * 110592, h1, b2s + i * 64,
                                             mbuf, h2);
    k_gemm1x1<4, 3><<<NTOT / 64, 256, 0, stream>>>(W3 + i * 4096, h2, b3s + i * 64,
                                                   mbuf, mcbuf, x, x);
  }
  // final 1x1x1 conv CH=64 -> PRED=32, NCDHW into d_out
  k_gemm1x1<2, 4><<<NTOT / 128, 256, 0, stream>>>(Wf, x, bf, nullptr, nullptr,
                                                  nullptr, out);
}
